// PtrNet_80839874445813
// MI455X (gfx1250) — matrix operand order repaired and checked by reading, not compiled, not run on hardware
//
#include <hip/hip_runtime.h>
#include <cstdint>

#define BB 512
#define SS 128
#define DD 2
#define HH 512
#define H4 2048

typedef __bf16 bf16_t;
typedef bf16_t v16bf __attribute__((ext_vector_type(16)));
typedef bf16_t v8bf  __attribute__((ext_vector_type(8)));
typedef float  v8f   __attribute__((ext_vector_type(8)));

__device__ __forceinline__ float sigf(float x) { return 1.0f / (1.0f + expf(-x)); }

// ---------------- utility kernels ----------------
__global__ void fill_u32(uint32_t* p, uint32_t v, int n) {
  int i = blockIdx.x * blockDim.x + threadIdx.x;
  if (i < n) p[i] = v;
}

__global__ void cvt_f32_bf16(const float* __restrict__ s, bf16_t* __restrict__ d, int n) {
  int i = blockIdx.x * blockDim.x + threadIdx.x;
  if (i < n) d[i] = (bf16_t)s[i];
}

// ---------------- WMMA GEMM: Z[M,N] = X[M,K] (bf16 rowmajor) * W[N,K]^T (bf16 rowmajor) + bias ----
// Each wave computes a 32(M) x 64(N) tile: every B fragment feeds 2 WMMAs, halving
// L2 weight traffic vs a 16x64 tile; A fragments feed 4 WMMAs each.
__global__ __launch_bounds__(256)
void gemm_xwt(const bf16_t* __restrict__ X, const bf16_t* __restrict__ W,
              const float* __restrict__ bias,
              float* __restrict__ Z, bf16_t* __restrict__ Zbf,
              int M, int N, int K)
{
  const int lane = threadIdx.x & 31;
  const int wave = (blockIdx.x * blockDim.x + threadIdx.x) >> 5;
  const int ntiles = N >> 6;
  const int total = (M >> 5) * ntiles;
  if (wave >= total) return;
  const int m0 = (wave / ntiles) << 5;
  const int n0 = (wave % ntiles) << 6;

  v8f acc[2][4] = {};

  const int ashift = (lane >> 4) << 3;   // 0 or 8  (K offset for A halves)
  const int bshift = (lane >> 4) << 3;   // 0 or 8  (K offset for B halves)
  const bf16_t* __restrict__ xr0 = X + (size_t)(m0 + (lane & 15)) * K;
  const bf16_t* __restrict__ xr1 = xr0 + (size_t)16 * K;
  const bf16_t* __restrict__ wc0 = W + (size_t)(n0 + (lane & 15)) * K + bshift;

  for (int kb = 0; kb < K; kb += 32) {
    if (kb + 32 < K) {
      // WGP-scope prefetch (locality=3): pull next k-block into the near cache
      // while the current clause of loads + WMMAs drains.  (SYS-scope would
      // stop at GL2, where the weights already live.)
      __builtin_prefetch(xr0 + kb + 32 + ashift, 0, 3);
      __builtin_prefetch(xr1 + kb + 32 + ashift, 0, 3);
      __builtin_prefetch(wc0 + kb + 32, 0, 3);
    }
    union { v16bf v; v8bf h[2]; } a0, a1;
    a0.h[0] = *(const v8bf*)(xr0 + kb + ashift);       // K = kb+{0..7}  (+8 upper lanes)
    a0.h[1] = *(const v8bf*)(xr0 + kb + ashift + 16);  // K = kb+{16..23}(+8 upper lanes)
    a1.h[0] = *(const v8bf*)(xr1 + kb + ashift);
    a1.h[1] = *(const v8bf*)(xr1 + kb + ashift + 16);
#pragma unroll
    for (int j = 0; j < 4; ++j) {
      const bf16_t* wp = W + (size_t)(n0 + j * 16 + (lane & 15)) * K + kb + bshift;
      union { v16bf v; v8bf h[2]; } bu;
      bu.h[0] = *(const v8bf*)wp;                      // K = kb+{0..7}  (+8 upper lanes)
      bu.h[1] = *(const v8bf*)(wp + 16);               // K = kb+{16..23}(+8 upper lanes)
      v16bf b = bu.v;
      acc[0][j] = __builtin_amdgcn_wmma_f32_16x16x32_bf16(false, a0.v, false, b,
                                                          (short)0, acc[0][j], false, false);
      acc[1][j] = __builtin_amdgcn_wmma_f32_16x16x32_bf16(false, a1.v, false, b,
                                                          (short)0, acc[1][j], false, false);
    }
  }

  const int cbase = n0 + (lane & 15);
#pragma unroll
  for (int t = 0; t < 2; ++t) {
    const int rbase = m0 + t * 16 + ((lane >> 4) << 3);
#pragma unroll
    for (int j = 0; j < 4; ++j) {
      const int col = cbase + j * 16;
      const float bv = bias ? bias[col] : 0.0f;
#pragma unroll
      for (int r = 0; r < 8; ++r) {
        float v = acc[t][j][r] + bv;
        size_t o = (size_t)(rbase + r) * N + col;
        if (Z)   Z[o]   = v;
        if (Zbf) Zbf[o] = (bf16_t)v;
      }
    }
  }
}

// ---------------- LSTM gate nonlinearity + state update ----------------
// G: [B,4H] = h @ Whh^T (f32, from WMMA).  x-contribution (K=D=2) + biases folded here.
__global__ __launch_bounds__(256)
void lstm_update(const float* __restrict__ G,
                 const float* __restrict__ xin, int xstride,
                 const float* __restrict__ Wih,
                 const float* __restrict__ bih, const float* __restrict__ bhh,
                 float* __restrict__ c, bf16_t* __restrict__ hbf,
                 bf16_t* __restrict__ extra, int extra_stride)
{
  const int idx = blockIdx.x * blockDim.x + threadIdx.x;  // b*H + n
  const int b = idx >> 9, n = idx & (HH - 1);
  const float x0 = xin[(size_t)b * xstride + 0];
  const float x1 = xin[(size_t)b * xstride + 1];
  const float* g = G + (size_t)b * H4;
  const int ni = n, nf = HH + n, ng = 2 * HH + n, no = 3 * HH + n;
  float zi = g[ni] + x0 * Wih[ni * DD] + x1 * Wih[ni * DD + 1] + bih[ni] + bhh[ni];
  float zf = g[nf] + x0 * Wih[nf * DD] + x1 * Wih[nf * DD + 1] + bih[nf] + bhh[nf];
  float zg = g[ng] + x0 * Wih[ng * DD] + x1 * Wih[ng * DD + 1] + bih[ng] + bhh[ng];
  float zo = g[no] + x0 * Wih[no * DD] + x1 * Wih[no * DD + 1] + bih[no] + bhh[no];
  float cn = sigf(zf) * c[idx] + sigf(zi) * tanhf(zg);
  float hn = sigf(zo) * tanhf(cn);
  c[idx] = cn;
  hbf[idx] = (bf16_t)hn;
  if (extra) extra[(size_t)b * extra_stride + n] = (bf16_t)hn;
}

// ---------------- attention logits: u[b,s] = v_b + sum_h v_w[h]*tanh(W1e[s,b,h] + q[b,h]) ----
__global__ __launch_bounds__(256)
void attn_u(const float* __restrict__ W1e, const float* __restrict__ q,
            const float* __restrict__ vw, const float* __restrict__ vb,
            float* __restrict__ ubuf)
{
  const int lane = threadIdx.x & 31;
  const int wave = (blockIdx.x * blockDim.x + threadIdx.x) >> 5;  // b*S + s
  const int b = wave >> 7;
  const int s = wave & (SS - 1);
  const float* w1 = W1e + ((size_t)s * BB + b) * HH;
  const float* qb = q + (size_t)b * HH;
  float acc = 0.f;
#pragma unroll
  for (int i = 0; i < 16; ++i) {
    int h = lane * 16 + i;
    acc += vw[h] * tanhf(w1[h] + qb[h]);
  }
  for (int off = 16; off; off >>= 1) acc += __shfl_down(acc, off);
  if (lane == 0) ubuf[(size_t)b * SS + s] = acc + vb[0];
}

// ---------------- softmax over S (wave per batch row) + argmax gather of next input ----
__global__ __launch_bounds__(256)
void softmax_sel(const float* __restrict__ ubuf, const float* __restrict__ inp,
                 float* __restrict__ out, float* __restrict__ cur, int step)
{
  const int lane = threadIdx.x & 31;
  const int b = (blockIdx.x * blockDim.x + threadIdx.x) >> 5;
  float uv[4];
  float bv = -3.4e38f; int bi = SS;
#pragma unroll
  for (int t = 0; t < 4; ++t) {
    int s = lane + 32 * t;
    uv[t] = ubuf[(size_t)b * SS + s];
    if (uv[t] > bv) { bv = uv[t]; bi = s; }
  }
  for (int off = 16; off; off >>= 1) {
    float ov = __shfl_down(bv, off);
    int   oi = __shfl_down(bi, off);
    if (ov > bv || (ov == bv && oi < bi)) { bv = ov; bi = oi; }  // first-max tiebreak
  }
  const float mx = __shfl(bv, 0);
  const int chosen = __shfl(bi, 0);
  float e[4], sum = 0.f;
#pragma unroll
  for (int t = 0; t < 4; ++t) { e[t] = expf(uv[t] - mx); sum += e[t]; }
  for (int off = 16; off; off >>= 1) sum += __shfl_down(sum, off);
  sum = __shfl(sum, 0);
  const float inv = 1.0f / sum;
  float* ob = out + ((size_t)b * SS + step) * SS;
#pragma unroll
  for (int t = 0; t < 4; ++t) ob[lane + 32 * t] = e[t] * inv;
  if (lane == 0) {
    cur[b * DD + 0] = inp[((size_t)b * SS + chosen) * DD + 0];
    cur[b * DD + 1] = inp[((size_t)b * SS + chosen) * DD + 1];
  }
}

// ---------------- ctx[b,h] = sum_s a[b,s] * enc_out[s,b,h]  (enc_out L2-resident bf16) ----
__global__ __launch_bounds__(256)
void ctx_kernel(const float* __restrict__ out, const bf16_t* __restrict__ enc,
                bf16_t* __restrict__ cat, int step)
{
  const int idx = blockIdx.x * blockDim.x + threadIdx.x;  // b*H + h
  const int b = idx >> 9, h = idx & (HH - 1);
  const float* a = out + ((size_t)b * SS + step) * SS;
  float acc = 0.f;
  for (int s = 0; s < SS; ++s)
    acc += a[s] * (float)enc[((size_t)s * BB + b) * HH + h];
  cat[(size_t)b * (2 * HH) + h] = (bf16_t)acc;
}

// ---------------- host orchestration ----------------
extern "C" void kernel_launch(void* const* d_in, const int* in_sizes, int n_in,
                              void* d_out, int out_size, void* d_ws, size_t ws_size,
                              hipStream_t stream)
{
  (void)in_sizes; (void)n_in; (void)out_size; (void)ws_size;
  const float* inp    = (const float*)d_in[0];
  const float* encWih = (const float*)d_in[1];
  const float* encWhh = (const float*)d_in[2];
  const float* encBih = (const float*)d_in[3];
  const float* encBhh = (const float*)d_in[4];
  const float* decWih = (const float*)d_in[5];
  const float* decWhh = (const float*)d_in[6];
  const float* decBih = (const float*)d_in[7];
  const float* decBhh = (const float*)d_in[8];
  const float* W1     = (const float*)d_in[9];
  const float* b1     = (const float*)d_in[10];
  const float* W2     = (const float*)d_in[11];
  const float* b2     = (const float*)d_in[12];
  const float* vw     = (const float*)d_in[13];
  const float* vb     = (const float*)d_in[14];
  const float* combW  = (const float*)d_in[15];
  const float* combB  = (const float*)d_in[16];
  float* out = (float*)d_out;

  char* base = (char*)d_ws;
  size_t off = 0;
  auto carve = [&](size_t bytes) -> void* {
    void* p = base + off;
    off = (off + bytes + 255) & ~(size_t)255;
    return p;
  };
  bf16_t* encWhh_bf = (bf16_t*)carve((size_t)H4 * HH * 2);
  bf16_t* decWhh_bf = (bf16_t*)carve((size_t)H4 * HH * 2);
  bf16_t* W1_bf     = (bf16_t*)carve((size_t)HH * HH * 2);
  bf16_t* W2_bf     = (bf16_t*)carve((size_t)HH * HH * 2);
  bf16_t* combW_bf  = (bf16_t*)carve((size_t)HH * 2 * HH * 2);
  bf16_t* hbf       = (bf16_t*)carve((size_t)BB * HH * 2);
  bf16_t* cat_bf    = (bf16_t*)carve((size_t)BB * 2 * HH * 2);
  bf16_t* encout_bf = (bf16_t*)carve((size_t)SS * BB * HH * 2);   // 64 MB, L2-resident
  float*  cbuf      = (float*)carve((size_t)BB * HH * 4);
  float*  G         = (float*)carve((size_t)BB * H4 * 4);
  float*  W1e       = (float*)carve((size_t)SS * BB * HH * 4);    // 128 MB
  float*  q         = (float*)carve((size_t)BB * HH * 4);
  float*  ubuf      = (float*)carve((size_t)BB * SS * 4);
  float*  cur      = (float*)carve((size_t)BB * DD * 4);

  auto gemm = [&](const bf16_t* X, const bf16_t* Wm, const float* bias,
                  float* Z, bf16_t* Zbf, int M, int N, int K) {
    int waves = (M >> 5) * (N >> 6);
    int blocks = (waves + 7) >> 3;
    gemm_xwt<<<blocks, 256, 0, stream>>>(X, Wm, bias, Z, Zbf, M, N, K);
  };
  auto cvt = [&](const float* s, bf16_t* d, int n) {
    cvt_f32_bf16<<<(n + 255) / 256, 256, 0, stream>>>(s, d, n);
  };

  // One-time weight conversion (per call; deterministic)
  cvt(encWhh, encWhh_bf, H4 * HH);
  cvt(decWhh, decWhh_bf, H4 * HH);
  cvt(W1, W1_bf, HH * HH);
  cvt(W2, W2_bf, HH * HH);
  cvt(combW, combW_bf, HH * 2 * HH);

  // State init: h=c=0, cur=-1
  fill_u32<<<(BB * HH + 255) / 256, 256, 0, stream>>>((uint32_t*)cbuf, 0u, BB * HH);
  fill_u32<<<(BB * HH / 2 + 255) / 256, 256, 0, stream>>>((uint32_t*)hbf, 0u, BB * HH / 2);
  union { float f; uint32_t u; } neg1; neg1.f = -1.0f;
  fill_u32<<<(BB * DD + 255) / 256, 256, 0, stream>>>((uint32_t*)cur, neg1.u, BB * DD);

  // ---- encoder ----
  for (int s = 0; s < SS; ++s) {
    gemm(hbf, encWhh_bf, nullptr, G, nullptr, BB, H4, HH);
    lstm_update<<<BB * HH / 256, 256, 0, stream>>>(
        G, inp + s * DD, SS * DD, encWih, encBih, encBhh,
        cbuf, hbf, encout_bf + (size_t)s * BB * HH, HH);
  }

  // ---- W1e = enc_out @ W1^T + b1  ([S*B, H] x [H, H]) ----
  gemm(encout_bf, W1_bf, b1, W1e, nullptr, SS * BB, HH, HH);

  // ---- pointer decoder ----
  for (int s = 0; s < SS; ++s) {
    gemm(hbf, decWhh_bf, nullptr, G, nullptr, BB, H4, HH);
    lstm_update<<<BB * HH / 256, 256, 0, stream>>>(
        G, cur, DD, decWih, decBih, decBhh,
        cbuf, hbf, cat_bf + HH, 2 * HH);                 // writes h-half of concat
    gemm(hbf, W2_bf, b2, q, nullptr, BB, HH, HH);
    attn_u<<<BB * SS / 8, 256, 0, stream>>>(W1e, q, vw, vb, ubuf);
    softmax_sel<<<BB / 8, 256, 0, stream>>>(ubuf, inp, out, cur, s);
    ctx_kernel<<<BB * HH / 256, 256, 0, stream>>>(out, encout_bf, cat_bf, s);
    gemm(cat_bf, combW_bf, combB, nullptr, hbf, BB, HH, 2 * HH);  // new h (bf16)
  }
}
